// LearnableSortingNetwork_37220186587467
// MI455X (gfx1250) — matrix operand order repaired and checked by reading, not compiled, not run on hardware
//
#include <hip/hip_runtime.h>
#include <hip/hip_bf16.h>

// ---------------------------------------------------------------------------
// LearnableSortingNetwork for MI455X (gfx1250, wave32).
//   K1: scores + argsort + 50 Sinkhorn iterations per batch (1 block each),
//       and (if ws allows) a bf16 copy of h written during the scores stream.
//   K2: Z = bf16(diag(u) Q diag(v)) @ bf16(h) via v_wmma_f32_16x16x32_bf16.
//       bf16 path stages B-tiles with the Tensor Data Mover (tensor_load_to_lds,
//       double-buffered, s_wait_tensorcnt); fp32 path is the fallback.
// The reference's convergence flag can never trigger (Sinkhorn col sums -> 1,
// delta -> 1-1/N ~ 0.996 > 1e-3), so exactly 50 iterations run and batches
// are independent.
// ---------------------------------------------------------------------------

typedef __attribute__((ext_vector_type(16))) __bf16 v16bf;
typedef __attribute__((ext_vector_type(8)))  __bf16 v8bf;
typedef __attribute__((ext_vector_type(8)))  float  v8f;
typedef __attribute__((ext_vector_type(4)))  unsigned int u32x4;
typedef __attribute__((ext_vector_type(8)))  unsigned int u32x8;

#define NN   256
#define DD   768
#define ITER 50
#define KB   32
#define LDP  48
#define NBLK 128

__device__ __forceinline__ int iabs(int x) { return x < 0 ? -x : x; }

template <int N>
__device__ __forceinline__ void wait_tensorcnt() {
#if __has_builtin(__builtin_amdgcn_s_wait_tensorcnt)
  __builtin_amdgcn_s_wait_tensorcnt((short)N);
#else
  if (N == 0) asm volatile("s_wait_tensorcnt 0x0" ::: "memory");
  else        asm volatile("s_wait_tensorcnt 0x1" ::: "memory");
#endif
}

// Issue one TDM 2-D tile load: 32 rows (stride 768 bf16) x 128 bf16 -> LDS.
// D# built per CDNA5 ISA (2-group form).
__device__ __forceinline__ void tdm_issue_tile(const __bf16* gsrc,
                                               unsigned lds_off) {
  unsigned long long ga = (unsigned long long)(uintptr_t)gsrc;
  u32x4 g0;
  g0.x = 1u;                                   // count=1, is_restore=0
  g0.y = lds_off;                              // lds_addr
  g0.z = (unsigned)(ga & 0xFFFFFFFFu);         // global_addr[31:0]
  g0.w = (unsigned)((ga >> 32) & 0x1FFFFFFu)   // global_addr[56:32]
       | (2u << 30);                           // type=2 ("image")
  u32x8 g1;
  g1.s0 = (1u << 16);                          // data_size=1 -> 2 bytes
  g1.s1 = (DD & 0xFFFFu) << 16;                // tensor_dim0[15:0] @ bits63:48
  g1.s2 = ((DD >> 16) & 0xFFFFu)               // tensor_dim0[31:16]
        | ((NN & 0xFFFFu) << 16);              // tensor_dim1[15:0]
  g1.s3 = ((NN >> 16) & 0xFFFFu)               // tensor_dim1[31:16]
        | ((unsigned)NBLK << 16);              // tile_dim0 = 128
  g1.s4 = (unsigned)KB;                        // tile_dim1 = 32, tile_dim2 = 0
  g1.s5 = (unsigned)DD;                        // tensor_dim0_stride[31:0]
  g1.s6 = 0u;                                  // stride hi / dim1_stride lo
  g1.s7 = 0u;
  asm volatile("tensor_load_to_lds %0, %1" :: "s"(g0), "s"(g1) : "memory");
}

// ===========================================================================
// Kernel 1: one block per batch. scores -> argsort -> Sinkhorn -> u,v,s to ws
// Optionally writes hb = bf16(h) while streaming h for the scores.
// ===========================================================================
__global__ __launch_bounds__(256) void k_sinkhorn(
    const float* __restrict__ h, const float* __restrict__ w,
    const float* __restrict__ bias,
    float* __restrict__ u_g, float* __restrict__ v_g, int* __restrict__ s_g,
    __bf16* __restrict__ hb)
{
  __shared__ float w_s[DD];
  __shared__ float sc_s[NN];
  __shared__ int   s_s[NN];
  __shared__ float u_s[NN], v_s[NN], ek_s[NN];

  const int b    = blockIdx.x;
  const int tid  = threadIdx.x;
  const int wave = tid >> 5;
  const int lane = tid & 31;

  for (int i = tid; i < DD; i += 256) w_s[i] = w[i];
  ek_s[tid] = expf(-10.0f * (float)tid);   // exp(-k/EPS), EPS=0.1
  const float bb = bias[0];
  __syncthreads();

  // ---- scores: wave handles rows wave, wave+8, ... (coalesced float2 reads)
  for (int it = 0; it < 32; ++it) {
    const int r = wave + 8 * it;
    const size_t rowoff = ((size_t)b * NN + r) * DD;
    const float2* hp = (const float2*)(h + rowoff);
    unsigned* hbrow = hb ? (unsigned*)(hb + rowoff) : nullptr;
    float p = 0.0f;
    for (int c = lane; c < DD / 2; c += 32) {
      float2 hv = hp[c];
      p = fmaf(hv.x, w_s[2 * c], fmaf(hv.y, w_s[2 * c + 1], p));
      if (hbrow) {
        unsigned lo = __builtin_bit_cast(unsigned short, (__bf16)hv.x);
        unsigned hi = __builtin_bit_cast(unsigned short, (__bf16)hv.y);
        hbrow[c] = lo | (hi << 16);
      }
    }
    #pragma unroll
    for (int off = 16; off; off >>= 1) p += __shfl_xor(p, off, 32);
    if (lane == 0) sc_s[r] = p + bb;
  }
  __syncthreads();

  // ---- stable ascending argsort via rank counting
  {
    const float xi = sc_s[tid];
    int rank = 0;
    for (int k = 0; k < NN; ++k) {
      const float xk = sc_s[k];
      rank += (int)((xk < xi) || (xk == xi && k < tid));
    }
    s_s[rank] = tid;          // s_b[rank_i] = i
  }
  u_s[tid] = 1.0f;
  v_s[tid] = 1.0f;
  __syncthreads();

  // ---- 50 Sinkhorn iterations; Q[i][j] = ek[|i - s[j]|]
  const int sj = s_s[tid];
  for (int it = 0; it < ITER; ++it) {
    float av = 0.0f;                       // (Q^T u)[tid]
    for (int i = 0; i < NN; ++i) av += ek_s[iabs(i - sj)] * u_s[i];
    v_s[tid] = 1.0f / (av + 1e-8f);
    __syncthreads();
    float au = 0.0f;                       // (Q v)[tid]
    for (int j = 0; j < NN; ++j) au += ek_s[iabs(tid - s_s[j])] * v_s[j];
    u_s[tid] = 1.0f / (au + 1e-8f);
    __syncthreads();
  }

  u_g[b * NN + tid] = u_s[tid];
  v_g[b * NN + tid] = v_s[tid];
  s_g[b * NN + tid] = s_s[tid];
}

// ===========================================================================
// Kernel 2: Z[b] = bf16(Qn[b]) @ bf16(h[b]), Qn generated on the fly.
// Block = (batch, 128-col slab), 8 waves; wave w -> rows 32w..32w+31.
// USE_HB: B tiles DMA'd by the TDM into a double-buffered raw LDS region,
//         then transposed LDS->LDS into the padded fragment layout.
// !USE_HB: B tiles read as fp32 from h and converted.
// ===========================================================================
template <bool USE_HB>
__global__ __launch_bounds__(256) void k_wmma_gemm(
    const float* __restrict__ h, const __bf16* __restrict__ hb,
    const float* __restrict__ u_g, const float* __restrict__ v_g,
    const int* __restrict__ s_g, __bf16* __restrict__ Z)
{
  __shared__ float u_s[NN], v_s[NN], ek_s[NN];
  __shared__ int   s_s[NN];
  __shared__ __attribute__((aligned(16))) __bf16 A_s[NN * LDP];        // 24 KB
  __shared__ __attribute__((aligned(16))) __bf16 Bt_s[NBLK * LDP];     // 12 KB
  __shared__ __attribute__((aligned(16))) __bf16 raw[2][KB * NBLK];    // 16 KB

  const int b    = blockIdx.x / (DD / NBLK);
  const int n0   = (blockIdx.x % (DD / NBLK)) * NBLK;
  const int tid  = threadIdx.x;
  const int wave = tid >> 5;
  const int lane = tid & 31;
  const int g    = lane >> 4;   // half-wave
  const int lhi  = lane & 15;

  u_s[tid]  = u_g[b * NN + tid];
  v_s[tid]  = v_g[b * NN + tid];
  s_s[tid]  = s_g[b * NN + tid];
  ek_s[tid] = expf(-10.0f * (float)tid);
  __syncthreads();

  v8f acc[2][8];
  #pragma unroll
  for (int mt = 0; mt < 2; ++mt)
    #pragma unroll
    for (int t = 0; t < 8; ++t) acc[mt][t] = (v8f)0.0f;

  const float ui = u_s[tid];
  const __bf16* hb_base = USE_HB ? hb + ((size_t)b * NN) * DD + n0 : nullptr;

  if constexpr (USE_HB) {
    if (wave == 0)
      tdm_issue_tile(hb_base, (unsigned)(uintptr_t)&raw[0][0]);
  }

  for (int kk = 0; kk < NN / KB; ++kk) {
    const int K0 = kk * KB;

    if constexpr (USE_HB) {
      if (wave == 0) {
        if (kk + 1 < NN / KB)
          tdm_issue_tile(hb_base + (size_t)(K0 + KB) * DD,
                         (unsigned)(uintptr_t)&raw[(kk + 1) & 1][0]);
        if (kk + 1 < NN / KB) wait_tensorcnt<1>();
        else                  wait_tensorcnt<0>();
      }
      __syncthreads();   // raw[kk&1] visible to all waves
    }

    // ---- stage A tile: A[i][k] = bf16(u[i] * ek[|i - s[K0+k]|] * v[K0+k])
    {
      const int i = tid;
      #pragma unroll 8
      for (int k = 0; k < KB; ++k) {
        const int j = K0 + k;
        A_s[i * LDP + k] = (__bf16)(ui * ek_s[iabs(i - s_s[j])] * v_s[j]);
      }
    }
    // ---- stage B^T tile: Bt[n][k] = bf16(h[b][K0+k][n0+n])
    if constexpr (USE_HB) {
      const int buf = kk & 1;
      #pragma unroll
      for (int c0 = 0; c0 < 2; ++c0) {
        const int c  = tid + 256 * c0;      // 0..511
        const int k  = c >> 4;              // 0..31
        const int n8 = (c & 15) << 3;       // 0,8,...,120
        v8bf vv = *(const v8bf*)&raw[buf][k * NBLK + n8];
        #pragma unroll
        for (int i = 0; i < 8; ++i) Bt_s[(n8 + i) * LDP + k] = vv[i];
      }
    } else {
      #pragma unroll 4
      for (int r = 0; r < (KB * NBLK) / 256; ++r) {
        const int lin = tid + 256 * r;
        const int k = lin >> 7;
        const int n = lin & (NBLK - 1);
        const float hv = h[((size_t)b * NN + K0 + k) * DD + n0 + n];
        Bt_s[n * LDP + k] = (__bf16)hv;
      }
    }
    __syncthreads();

    // ---- compute: per wave, 2 A fragments x 8 B fragments
    #pragma unroll
    for (int mt = 0; mt < 2; ++mt) {
      const int m = wave * 32 + mt * 16 + lhi;
      // A frag (16x32): lane m; elems 0..7 = K 8g..8g+7, 8..15 = K 16+8g..
      const __bf16* arow = &A_s[m * LDP];
      v8bf alo = *(const v8bf*)(arow + 8 * g);
      v8bf ahi = *(const v8bf*)(arow + 16 + 8 * g);
      v16bf afr = __builtin_shufflevector(alo, ahi,
          0, 1, 2, 3, 4, 5, 6, 7, 8, 9, 10, 11, 12, 13, 14, 15);
      #pragma unroll
      for (int t = 0; t < 8; ++t) {
        // B frag (32x16): lane n = 16t+lhi; elems 0..7 = K 8g..8g+7, 8..15 = K 16+8g..
        const __bf16* brow = &Bt_s[(16 * t + lhi) * LDP];
        v8bf blo = *(const v8bf*)(brow + 8 * g);
        v8bf bhi = *(const v8bf*)(brow + 16 + 8 * g);
        v16bf bfr = __builtin_shufflevector(blo, bhi,
            0, 1, 2, 3, 4, 5, 6, 7, 8, 9, 10, 11, 12, 13, 14, 15);
        acc[mt][t] = __builtin_amdgcn_wmma_f32_16x16x32_bf16(
            false, afr, false, bfr, (short)0, acc[mt][t], false, false);
      }
    }
    __syncthreads();
  }

  // ---- epilogue: D layout -> row = m0 + r + 8g, col = n0 + 16t + lhi
  #pragma unroll
  for (int mt = 0; mt < 2; ++mt) {
    #pragma unroll
    for (int t = 0; t < 8; ++t) {
      const v8f c = acc[mt][t];
      const int col = n0 + 16 * t + lhi;
      #pragma unroll
      for (int r = 0; r < 8; ++r) {
        const int row = wave * 32 + mt * 16 + r + 8 * g;
        Z[((size_t)b * NN + row) * DD + col] = (__bf16)c[r];
      }
    }
  }
}

// ===========================================================================
extern "C" void kernel_launch(void* const* d_in, const int* in_sizes, int n_in,
                              void* d_out, int out_size, void* d_ws, size_t ws_size,
                              hipStream_t stream) {
  const float* h    = (const float*)d_in[0];
  const float* w    = (const float*)d_in[1];
  const float* bias = (const float*)d_in[2];
  const int B = in_sizes[0] / (NN * DD);

  float* u_g = (float*)d_ws;                 // B*256 f32
  float* v_g = u_g + (size_t)B * NN;         // B*256 f32
  int*   s_g = (int*)(v_g + (size_t)B * NN); // B*256 i32
  const size_t base_bytes = 3u * (size_t)B * NN * 4u;      // ~196 KB
  const size_t hb_bytes   = (size_t)B * NN * DD * 2u;      // ~25 MB

  const bool use_hb = ws_size >= base_bytes + hb_bytes;
  __bf16* hb = use_hb ? (__bf16*)((char*)d_ws + base_bytes) : nullptr;

  k_sinkhorn<<<B, 256, 0, stream>>>(h, w, bias, u_g, v_g, s_g, hb);
  if (use_hb)
    k_wmma_gemm<true><<<B * (DD / NBLK), 256, 0, stream>>>(
        h, hb, u_g, v_g, s_g, (__bf16*)d_out);
  else
    k_wmma_gemm<false><<<B * (DD / NBLK), 256, 0, stream>>>(
        h, nullptr, u_g, v_g, s_g, (__bf16*)d_out);
}
